// FusedGeometricRound_8186207666890
// MI455X (gfx1250) — matrix operand order repaired and checked by reading, not compiled, not run on hardware
//
#include <hip/hip_runtime.h>
#include <hip/hip_bf16.h>
#include <stdint.h>

// ---------------------------------------------------------------------------
// FusedGeometric: Cl(3,0) geometric product + gate-mix + LayerNorm + SwiGLU FFN
// Compute-bound on GEMMs -> bf16 WMMA path, LDS-staged A tiles (async-to-LDS),
// 1x4 register blocking per wave for A-fragment reuse.
// ---------------------------------------------------------------------------

typedef __attribute__((ext_vector_type(16))) __bf16 v16bf;
typedef __attribute__((ext_vector_type(8)))  __bf16 v8bf;
typedef __attribute__((ext_vector_type(8)))  float  v8f;
typedef __attribute__((ext_vector_type(4)))  int    v4i;

typedef __attribute__((address_space(1))) v4i gv4i;   // global int4
typedef __attribute__((address_space(3))) v4i lv4i;   // LDS int4

#define N_TOK   32768
#define D_BLADE 256
#define D_FFN   1024
#define M_ROWS  (N_TOK * 8)   // 262144 flat rows

// gfx1250 async global->LDS copy (ASYNCcnt-tracked); guarded so compilation
// can never break if the builtin is absent on this toolchain.
#if defined(__has_builtin)
# if __has_builtin(__builtin_amdgcn_global_load_async_to_lds_b128)
#  define ASYNC_LDS 1
# endif
# if __has_builtin(__builtin_amdgcn_s_wait_asynccnt)
#  define HAVE_WAIT_ASYNC 1
# endif
#endif

__device__ __forceinline__ void async_copy16(void* g, void* l) {
#ifdef ASYNC_LDS
    __builtin_amdgcn_global_load_async_to_lds_b128(
        (gv4i*)g, (lv4i*)l, 0, 0);
#else
    *(v8bf*)l = *(const v8bf*)g;
#endif
}

__device__ __forceinline__ void async_wait0() {
#ifdef ASYNC_LDS
# ifdef HAVE_WAIT_ASYNC
    __builtin_amdgcn_s_wait_asynccnt(0);
# else
    asm volatile("s_wait_asynccnt 0x0" ::: "memory");
# endif
#endif
}

__device__ __forceinline__ uint16_t f2bf(float f) {
    uint32_t u = __float_as_uint(f);
    uint32_t r = u + 0x7fffu + ((u >> 16) & 1u);   // round-to-nearest-even
    return (uint16_t)(r >> 16);
}

__device__ __forceinline__ float sigmoidf_(float t) {
    return 1.0f / (1.0f + __expf(-t));
}

// A-fragment, contiguous K order (matches B): lanes 0-15 hold K {0..15},
// lanes 16-31 hold K {16..31} of the 32-wide K block.
__device__ __forceinline__ v16bf a_frag(const __bf16* __restrict__ row,
                                        int kbase, int half) {
    int ka = kbase + half * 16;
    v8bf lo = *(const v8bf*)(row + ka);
    v8bf hi = *(const v8bf*)(row + ka + 8);
    v16bf a;
    #pragma unroll
    for (int i = 0; i < 8; ++i) { a[i] = lo[i]; a[8 + i] = hi[i]; }
    return a;
}

#define WMMA_BF16(A, B, C) \
    __builtin_amdgcn_wmma_f32_16x16x32_bf16(false, (A), false, (B), (short)0, (C), false, false)

// ---------------------------------------------------------------------------
// Kernel 1: geometric product + gate mix + LayerNorm -> bf16 flat activations.
// One block per token (256 threads = 8 waves, one wave per blade-row for LN).
// ---------------------------------------------------------------------------
__global__ __launch_bounds__(256) void geo_ln_kernel(
    const float* __restrict__ x, const float* __restrict__ iw,
    const float* __restrict__ gate, const float* __restrict__ lnw,
    const float* __restrict__ lnb, uint16_t* __restrict__ flatb)
{
    __shared__ float coef[64];        // sign(i,j) * sigmoid(iw[i*8+j])
    __shared__ float smem[8 * 256];   // mixed values for LN reduction
    __shared__ float red[16];         // per-blade {mu, rsigma}

    const int n = blockIdx.x;
    const int d = threadIdx.x;

    if (threadIdx.x < 64) {
        int i = threadIdx.x >> 3, j = threadIdx.x & 7;
        int s = 0, a = i >> 1;
        while (a) { s += __popc(a & j); a >>= 1; }
        float sg = (s & 1) ? -1.0f : 1.0f;
        coef[threadIdx.x] = sg * sigmoidf_(iw[threadIdx.x]);
    }
    __syncthreads();

    float xv[8];
    #pragma unroll
    for (int i = 0; i < 8; ++i)
        xv[i] = x[(size_t)n * 2048 + i * 256 + d];

    const float g = sigmoidf_(gate[0]);

    float mv[8];
    #pragma unroll
    for (int k = 0; k < 8; ++k) {
        float acc = 0.0f;
        #pragma unroll
        for (int i = 0; i < 8; ++i)
            acc = fmaf(coef[i * 8 + (i ^ k)] * xv[i], xv[i ^ k], acc);
        mv[k] = g * acc + (1.0f - g) * xv[k];
        smem[k * 256 + d] = mv[k];
    }
    __syncthreads();

    const int wid = threadIdx.x >> 5, lane = threadIdx.x & 31;
    {
        float s = 0.0f, s2 = 0.0f;
        #pragma unroll
        for (int j = 0; j < 8; ++j) {
            float v = smem[wid * 256 + lane + 32 * j];
            s += v; s2 = fmaf(v, v, s2);
        }
        #pragma unroll
        for (int o = 16; o > 0; o >>= 1) {
            s  += __shfl_xor(s,  o, 32);
            s2 += __shfl_xor(s2, o, 32);
        }
        if (lane == 0) {
            float mu  = s * (1.0f / 256.0f);
            float var = s2 * (1.0f / 256.0f) - mu * mu;
            red[wid * 2]     = mu;
            red[wid * 2 + 1] = rsqrtf(var + 1e-5f);
        }
    }
    __syncthreads();

    const float w0 = lnw[d], b0 = lnb[d];
    #pragma unroll
    for (int k = 0; k < 8; ++k) {
        float mu = red[k * 2], rs = red[k * 2 + 1];
        float v = (mv[k] - mu) * rs * w0 + b0;
        flatb[((size_t)n * 8 + k) * 256 + d] = f2bf(v);
    }
}

// ---------------------------------------------------------------------------
// Kernel 2: convert weights to bf16, transposed to [nCol][k] so each WMMA
// B-fragment is one contiguous 32B load per lane.
// ---------------------------------------------------------------------------
__global__ __launch_bounds__(256) void wconv_kernel(
    const float* __restrict__ wg, const float* __restrict__ wu,
    const float* __restrict__ wd,
    uint16_t* __restrict__ Bg, uint16_t* __restrict__ Bu,
    uint16_t* __restrict__ Bd)
{
    int idx = blockIdx.x * 256 + threadIdx.x;
    if (idx < 1024 * 256) {   // Bg/Bu: n<1024, k<256 ; src w[k*1024+n]
        int nn = idx >> 8, k = idx & 255;
        Bg[idx] = f2bf(wg[(size_t)k * 1024 + nn]);
        Bu[idx] = f2bf(wu[(size_t)k * 1024 + nn]);
    }
    if (idx < 256 * 1024) {   // Bd: n<256, k<1024 ; src wd[k*256+n]
        int nn = idx >> 10, k = idx & 1023;
        Bd[idx] = f2bf(wd[(size_t)k * 256 + nn]);
    }
}

// ---------------------------------------------------------------------------
// Kernel 3: [262144 x 256] @ [256 x (1024|1024)] gate+up, fused silu(g)*u.
// Block = 16 A rows staged once in LDS (8KB, async), 8 waves x (2 gate +
// 2 up) 16x16 tiles -> 4 WMMAs share one A fragment per K step.
// Grid: 16384 M-tiles x 4 column groups.
// ---------------------------------------------------------------------------
#define LDA1 264   // padded LDS row stride (bf16) to spread banks
__global__ __launch_bounds__(256) void gemm_swiglu_kernel(
    const uint16_t* __restrict__ A_, const uint16_t* __restrict__ Bg_,
    const uint16_t* __restrict__ Bu_, uint16_t* __restrict__ H)
{
    __shared__ __align__(16) uint16_t As[16 * LDA1];

    const __bf16* Bg = (const __bf16*)Bg_;
    const __bf16* Bu = (const __bf16*)Bu_;

    const int bm  = blockIdx.x >> 2;   // M tile (16 rows)
    const int bn  = blockIdx.x & 3;    // column group (256 gate cols)
    const int tid = threadIdx.x;
    const int wid = tid >> 5, lane = tid & 31;
    const int half = lane >> 4, l15 = lane & 15;

    // Stage 16x256 bf16 A tile -> LDS (512 chunks of 16B, 2 per thread).
    {
        const uint16_t* gbase = A_ + (size_t)bm * 16 * 256;
        #pragma unroll
        for (int t = 0; t < 2; ++t) {
            int task = tid + t * 256;
            int r = task >> 5, c = task & 31;
            async_copy16((void*)(gbase + r * 256 + c * 8), As + r * LDA1 + c * 8);
        }
        async_wait0();
        __syncthreads();
    }

    const int gcol = bn * 256 + wid * 32;   // this wave's first gate column
    const __bf16* bg0 = Bg + (size_t)(gcol + l15) * 256;
    const __bf16* bg1 = Bg + (size_t)(gcol + 16 + l15) * 256;
    const __bf16* bu0 = Bu + (size_t)(gcol + l15) * 256;
    const __bf16* bu1 = Bu + (size_t)(gcol + 16 + l15) * 256;
    const __bf16* arow = (const __bf16*)(As + l15 * LDA1);

    v8f accg0 = {}, accg1 = {}, accu0 = {}, accu1 = {};
    #pragma unroll
    for (int kb = 0; kb < 8; ++kb) {
        v16bf a = a_frag(arow, kb * 32, half);
        int ko = kb * 32 + half * 16;
        v16bf g0 = *(const v16bf*)(bg0 + ko);
        v16bf g1 = *(const v16bf*)(bg1 + ko);
        v16bf u0 = *(const v16bf*)(bu0 + ko);
        v16bf u1 = *(const v16bf*)(bu1 + ko);
        accg0 = WMMA_BF16(a, g0, accg0);
        accg1 = WMMA_BF16(a, g1, accg1);
        accu0 = WMMA_BF16(a, u0, accu0);
        accu1 = WMMA_BF16(a, u1, accu1);
    }

    const int c0 = gcol + l15, c1 = gcol + 16 + l15;
    #pragma unroll
    for (int r = 0; r < 8; ++r) {
        int row = bm * 16 + r + half * 8;
        float g0 = accg0[r], g1 = accg1[r];
        float h0 = g0 * (1.0f / (1.0f + __expf(-g0))) * accu0[r];
        float h1 = g1 * (1.0f / (1.0f + __expf(-g1))) * accu1[r];
        H[(size_t)row * D_FFN + c0] = f2bf(h0);
        H[(size_t)row * D_FFN + c1] = f2bf(h1);
    }
}

// ---------------------------------------------------------------------------
// Kernel 4: [262144 x 1024] @ [1024 x 256] + residual, f32 output.
// Block = 16 H rows staged once in LDS (32KB, async); 8 waves x 2 N-tiles
// cover all 256 output columns -> h read exactly once from memory.
// Grid: 16384 blocks.
// ---------------------------------------------------------------------------
#define LDA2 1032  // padded LDS row stride (bf16)
__global__ __launch_bounds__(256) void gemm_down_res_kernel(
    const uint16_t* __restrict__ H_, const uint16_t* __restrict__ Bd_,
    const float* __restrict__ x, float* __restrict__ out)
{
    __shared__ __align__(16) uint16_t As[16 * LDA2];

    const __bf16* Bd = (const __bf16*)Bd_;

    const int bm  = blockIdx.x;        // M tile (16 rows)
    const int tid = threadIdx.x;
    const int wid = tid >> 5, lane = tid & 31;
    const int half = lane >> 4, l15 = lane & 15;

    // Stage 16x1024 bf16 H tile -> LDS (2048 chunks of 16B, 8 per thread).
    {
        const uint16_t* gbase = H_ + (size_t)bm * 16 * 1024;
        #pragma unroll
        for (int t = 0; t < 8; ++t) {
            int task = tid + t * 256;
            int r = task >> 7, c = task & 127;
            async_copy16((void*)(gbase + r * 1024 + c * 8), As + r * LDA2 + c * 8);
        }
        async_wait0();
        __syncthreads();
    }

    const int n0 = wid * 32;           // this wave's first output column
    const __bf16* b0 = Bd + (size_t)(n0 + l15) * 1024;
    const __bf16* b1 = Bd + (size_t)(n0 + 16 + l15) * 1024;
    const __bf16* arow = (const __bf16*)(As + l15 * LDA2);

    v8f acc0 = {}, acc1 = {};
    #pragma unroll
    for (int kb = 0; kb < 32; ++kb) {
        v16bf a = a_frag(arow, kb * 32, half);
        int ko = kb * 32 + half * 16;
        v16bf bb0 = *(const v16bf*)(b0 + ko);
        v16bf bb1 = *(const v16bf*)(b1 + ko);
        acc0 = WMMA_BF16(a, bb0, acc0);
        acc1 = WMMA_BF16(a, bb1, acc1);
    }

    const int c0 = n0 + l15, c1 = n0 + 16 + l15;
    #pragma unroll
    for (int r = 0; r < 8; ++r) {
        size_t row = (size_t)bm * 16 + r + half * 8;
        out[row * 256 + c0] = x[row * 256 + c0] + acc0[r];
        out[row * 256 + c1] = x[row * 256 + c1] + acc1[r];
    }
}

// ---------------------------------------------------------------------------
extern "C" void kernel_launch(void* const* d_in, const int* in_sizes, int n_in,
                              void* d_out, int out_size, void* d_ws, size_t ws_size,
                              hipStream_t stream) {
    const float* x    = (const float*)d_in[0];   // (32768, 8, 256)
    const float* iw   = (const float*)d_in[1];   // (64,)
    const float* gate = (const float*)d_in[2];   // scalar
    const float* lnw  = (const float*)d_in[3];   // (256,)
    const float* lnb  = (const float*)d_in[4];   // (256,)
    const float* wg   = (const float*)d_in[5];   // (256, 1024)
    const float* wu   = (const float*)d_in[6];   // (256, 1024)
    const float* wd   = (const float*)d_in[7];   // (1024, 256)
    float* out        = (float*)d_out;

    // Workspace layout (bf16 as uint16):
    //   flat : M_ROWS*256   (134 MB)
    //   Bg   : 1024*256
    //   Bu   : 1024*256
    //   Bd   : 256*1024
    //   H    : M_ROWS*1024  (537 MB)
    uint16_t* flatb = (uint16_t*)d_ws;
    uint16_t* Bg    = flatb + (size_t)M_ROWS * 256;
    uint16_t* Bu    = Bg + (size_t)1024 * 256;
    uint16_t* Bd    = Bu + (size_t)1024 * 256;
    uint16_t* H     = Bd + (size_t)256 * 1024;

    geo_ln_kernel<<<N_TOK, 256, 0, stream>>>(x, iw, gate, lnw, lnb, flatb);
    wconv_kernel<<<1024, 256, 0, stream>>>(wg, wu, wd, Bg, Bu, Bd);

    // (M/16) M-tiles x 4 column groups
    gemm_swiglu_kernel<<<65536, 256, 0, stream>>>(flatb, Bg, Bu, H);

    // (M/16) blocks; each covers all 256 output columns
    gemm_down_res_kernel<<<16384, 256, 0, stream>>>(H, Bd, x, out);
}
